// VectorQuantizer_69252052681260
// MI455X (gfx1250) — matrix operand order repaired and checked by reading, compiled, not run on hardware
//
#include <hip/hip_runtime.h>
#include <math.h>

typedef __attribute__((ext_vector_type(16))) __bf16 v16bf;
typedef __attribute__((ext_vector_type(8)))  float  v8f;
typedef __attribute__((ext_vector_type(2)))  float  v2f;
typedef __attribute__((ext_vector_type(4)))  unsigned int v4u;
typedef __attribute__((ext_vector_type(8)))  int    v8i;
typedef __attribute__((ext_vector_type(4)))  int    v4i;

#define N_PTS   32768
#define CDIM    128
#define KCODES  8192
#define HWDIM   1024
#define OUT_TENSOR (32*128*32*32)   // 4194304
#define PERP_OFF (1 + OUT_TENSOR)   // 4194305
#define IDX_OFF  (2 + OUT_TENSOR)   // 4194306
#define NCHUNK   256                // 2 k-tiles (32 codes) per chunk
#define CHUNK_BYTES 16384           // 2 kt * (hi+lo) * 2048 bf16 * 2B

// ---------------- codebook prep: bf16 hi/lo split, interleaved WMMA-B-fragment order ----
// epk element layout: [kt][h(0=hi,1=lo)][cs][lane][e]  (4096 bf16 per kt)
__global__ void k_prep_emb(const float* __restrict__ emb, __bf16* __restrict__ epk) {
    int t  = blockIdx.x * 256 + threadIdx.x;     // 512 kt * 4 cs * 32 lanes
    int l  = t & 31;
    int cs = (t >> 5) & 3;
    int kt = t >> 7;
    int code = kt * 16 + (l & 15);
    int cb   = cs * 32 + ((l & 16) ? 16 : 0);
    const float* src = emb + (size_t)code * CDIM + cb;
    __bf16* dh = epk + (size_t)kt * 4096 + cs * 512 + l * 16;
    __bf16* dl = dh + 2048;
#pragma unroll
    for (int e = 0; e < 16; e++) {
        float v = src[e];
        __bf16 h = (__bf16)v;
        dh[e] = h;
        dl[e] = (__bf16)(v - (float)h);
    }
}

__global__ void k_norm(const float* __restrict__ emb, float* __restrict__ hn) {
    int k = blockIdx.x * 256 + threadIdx.x;
    if (k >= KCODES) return;
    const float4* r = (const float4*)(emb + (size_t)k * CDIM);
    float s = 0.f;
#pragma unroll 8
    for (int i = 0; i < 32; i++) { float4 v = r[i]; s += v.x*v.x + v.y*v.y + v.z*v.z + v.w*v.w; }
    hn[k] = 0.5f * s;
}

__global__ void k_zero(int* __restrict__ hist) {
    int k = blockIdx.x * 256 + threadIdx.x;
    if (k < KCODES) hist[k] = 0;
}

// ---------------- projection: flat = transpose(x) @ W^T + b,  exact fp32 WMMA -----------
__global__ void __launch_bounds__(256) k_linear(const float* __restrict__ x,
                                                const float* __restrict__ Wl,
                                                const float* __restrict__ bl,
                                                float* __restrict__ flat) {
    __shared__ float At[64 * 132];
    int tid  = threadIdx.x;
    int n0   = blockIdx.x * 64;
    int bimg = n0 >> 10, rem = n0 & 1023;
    const float* xb = x + (size_t)bimg * CDIM * HWDIM + rem;
    for (int i = tid; i < 64 * CDIM; i += 256) {
        int c = i >> 6, p = i & 63;
        At[p * 132 + c] = xb[(size_t)c * HWDIM + p];
    }
    __syncthreads();

    int w = tid >> 5, l = tid & 31, m = l & 15;
    int mr = (w & 3) * 16;
    int jhalf = w >> 2;
    bool hiL = (l >= 16);

    v2f af[32];
#pragma unroll
    for (int cs = 0; cs < 32; cs++) {
        int cb = cs * 4 + (hiL ? 2 : 0);
        const float* ap = &At[(mr + m) * 132 + cb];
        af[cs].x = ap[0]; af[cs].y = ap[1];
    }
    for (int jt = 0; jt < 4; jt++) {
        int j0 = jhalf * 64 + jt * 16;
        float bias = bl[j0 + m];
        v8f acc;
#pragma unroll
        for (int i = 0; i < 8; i++) acc[i] = bias;
#pragma unroll
        for (int cs = 0; cs < 32; cs++) {
            int cb = cs * 4 + (hiL ? 2 : 0);
            v2f bv = *(const v2f*)(Wl + (size_t)(j0 + m) * CDIM + cb);
            acc = __builtin_amdgcn_wmma_f32_16x16x4_f32(false, af[cs], false, bv,
                                                        (short)0, acc, false, false);
        }
#pragma unroll
        for (int i = 0; i < 8; i++) {
            int row = mr + (hiL ? 8 + i : i);
            flat[(size_t)(n0 + row) * CDIM + j0 + m] = acc[i];
        }
    }
}

// ---- TDM: 1-D contiguous 16KB copy (4096 x 4B) global -> LDS, per ISA D# layout --------
__device__ __forceinline__ void tdm_load_chunk(const void* gsrc, unsigned lds_off) {
    unsigned long long ga = (unsigned long long)gsrc;
    v4u g0;
    g0[0] = 1u;                                       // count=1, user mode, no gather
    g0[1] = lds_off;                                  // lds_addr
    g0[2] = (unsigned)ga;                             // global_addr[31:0]
    g0[3] = (unsigned)((ga >> 32) & 0x01FFFFFFu) | (2u << 30);   // addr[56:32] | type=2
    v8i g1;
    g1[0] = (int)(2u << 16);                          // wg_mask=0, data_size=2 (4B)
    g1[1] = (int)((4096u & 0xFFFFu) << 16);           // tensor_dim0[15:0]
    g1[2] = (int)(1u << 16);                          // tensor_dim0[31:16]=0 | tensor_dim1=1
    g1[3] = (int)(4096u << 16);                       // tensor_dim1 hi=0 | tile_dim0=4096
    g1[4] = 0;                                        // tile_dim1=0, tile_dim2=0
    g1[5] = 4096;                                     // tensor_dim0_stride lo
    g1[6] = 0;
    g1[7] = 0;
    v4i g2 = {0, 0, 0, 0};
    v4i g3 = {0, 0, 0, 0};
    v8i g4 = {0, 0, 0, 0, 0, 0, 0, 0};
    __builtin_amdgcn_tensor_load_to_lds(g0, g1, g2, g3, g4, 0);
}

// ---------------- argmax_k (z.e_k - 0.5||e_k||^2): bf16x3 WMMA + TDM-staged B -----------
__global__ void __launch_bounds__(256) k_argmin(const float* __restrict__ flat,
                                                const __bf16* __restrict__ epk,
                                                const float* __restrict__ hn,
                                                int* __restrict__ idx_out,
                                                float* __restrict__ idxf_out) {
    __shared__ __align__(128) char smem[65536];       // 3 x 16KB TDM buffers + (aliased) scratch
    unsigned lds_base = (unsigned)(unsigned long long)
        (__attribute__((address_space(3))) char*)smem;

    int tid = threadIdx.x;
    int w = tid >> 5, l = tid & 31, m = l & 15;
    bool hiL = (l >= 16);
    int n0 = blockIdx.x * 128;
    int p  = n0 + w * 16 + m;
    const float* fp = flat + (size_t)p * CDIM;

    // A fragments (16 rows x 128 K), hi/lo bf16, register-resident for the whole loop
    v16bf ahi[4], alo[4];
#pragma unroll
    for (int cs = 0; cs < 4; cs++) {
        int c1 = cs * 32 + (hiL ? 16 : 0);
        v8f r1 = *(const v8f*)(fp + c1);
        v8f r2 = *(const v8f*)(fp + c1 + 8);
#pragma unroll
        for (int e = 0; e < 8; e++) {
            float v = r1[e]; __bf16 h = (__bf16)v;
            ahi[cs][e] = h;      alo[cs][e] = (__bf16)(v - (float)h);
            v = r2[e];       h = (__bf16)v;
            ahi[cs][8 + e] = h;  alo[cs][8 + e] = (__bf16)(v - (float)h);
        }
    }

    float best[8]; int bidx[8];
#pragma unroll
    for (int i = 0; i < 8; i++) { best[i] = -3.4e38f; bidx[i] = 0; }

    const char* gsrc = (const char*)epk;
    if (w == 0) {                                     // pre-issue pipeline depth 2
        tdm_load_chunk(gsrc,                 lds_base);
        tdm_load_chunk(gsrc + CHUNK_BYTES,   lds_base + CHUNK_BYTES);
    }

    for (int i = 0; i < NCHUNK; i++) {
        if (w == 0) {
            if (i == NCHUNK - 1) __builtin_amdgcn_s_wait_tensorcnt(0);
            else                 __builtin_amdgcn_s_wait_tensorcnt(1);
        }
        __syncthreads();                              // chunk i resident; chunk i-1 fully consumed
        if (w == 0 && (i + 2) < NCHUNK) {
            int b = (i + 2) % 3;
            tdm_load_chunk(gsrc + (size_t)(i + 2) * CHUNK_BYTES, lds_base + b * CHUNK_BYTES);
        }
        const char* buf = smem + (i % 3) * CHUNK_BYTES;
#pragma unroll
        for (int ktl = 0; ktl < 2; ktl++) {
            int kt = i * 2 + ktl;
            float nh = -hn[kt * 16 + m];
            v8f acc;
#pragma unroll
            for (int q = 0; q < 8; q++) acc[q] = nh;
            const char* kb = buf + ktl * 8192 + l * 32;
#pragma unroll
            for (int cs = 0; cs < 4; cs++) {
                const v16bf bh = *(const v16bf*)(kb + cs * 1024);
                const v16bf bo = *(const v16bf*)(kb + cs * 1024 + 4096);
                acc = __builtin_amdgcn_wmma_f32_16x16x32_bf16(false, alo[cs], false, bh, (short)0, acc, false, false);
                acc = __builtin_amdgcn_wmma_f32_16x16x32_bf16(false, ahi[cs], false, bo, (short)0, acc, false, false);
                acc = __builtin_amdgcn_wmma_f32_16x16x32_bf16(false, ahi[cs], false, bh, (short)0, acc, false, false);
            }
            int kc = kt * 16 + m;
#pragma unroll
            for (int q = 0; q < 8; q++) {
                if (acc[q] > best[q]) { best[q] = acc[q]; bidx[q] = kc; }
            }
        }
    }

    __syncthreads();                                  // done with TDM buffers; alias as scratch
    float* sval = (float*)smem;                       // 8KB
    int*   sidx = (int*)(smem + 8192);                // 8KB
#pragma unroll
    for (int i = 0; i < 8; i++) {
        int r = hiL ? 8 + i : i;
        sval[(w * 16 + r) * 16 + m] = best[i];
        sidx[(w * 16 + r) * 16 + m] = bidx[i];
    }
    __syncthreads();
    if (l < 16) {
        float bv = sval[(w * 16 + l) * 16]; int bi = sidx[(w * 16 + l) * 16];
#pragma unroll
        for (int c = 1; c < 16; c++) {
            float v = sval[(w * 16 + l) * 16 + c]; int ii = sidx[(w * 16 + l) * 16 + c];
            if (v > bv || (v == bv && ii < bi)) { bv = v; bi = ii; }
        }
        int n = n0 + w * 16 + l;
        idx_out[n]  = bi;
        idxf_out[n] = (float)bi;
    }
}

// ---------------- gather codes -> BCHW output, SSE partials, histogram ------------------
__global__ void __launch_bounds__(256) k_gather(const float* __restrict__ flat,
                                                const float* __restrict__ emb,
                                                const int* __restrict__ idx,
                                                float* __restrict__ out,
                                                int* __restrict__ hist,
                                                float* __restrict__ part) {
    __shared__ float fl[64 * 129];
    __shared__ int   codes[64];
    __shared__ float red[256];
    int tid = threadIdx.x;
    int n0  = blockIdx.x * 64;
    if (tid < 64) codes[tid] = idx[n0 + tid];
    for (int i = tid; i < 64 * CDIM; i += 256) {
        int pp = i >> 7, cc = i & 127;
        fl[pp * 129 + cc] = flat[(size_t)n0 * CDIM + i];
    }
    __syncthreads();

    int pl   = tid & 63;
    int cofs = tid >> 6;
    int bimg = n0 >> 10, rem = n0 & 1023;
    float* ob = out + (size_t)bimg * CDIM * HWDIM + rem + pl;
    int code = codes[pl];
    const float* er = emb + (size_t)code * CDIM;
    float sse = 0.f;
#pragma unroll 4
    for (int cc = 0; cc < 32; cc++) {
        int c = cc * 4 + cofs;
        float q = er[c];
        float d = q - fl[pl * 129 + c];
        sse += d * d;
        ob[(size_t)c * HWDIM] = q;
    }
    if (tid < 64) atomicAdd(&hist[codes[tid]], 1);
    red[tid] = sse;
    __syncthreads();
    for (int s = 128; s > 0; s >>= 1) {
        if (tid < s) red[tid] += red[tid + s];
        __syncthreads();
    }
    if (tid == 0) part[blockIdx.x] = red[0];
}

// ---------------- finalize: loss + perplexity (deterministic fixed-order sums) ----------
__global__ void k_final(const int* __restrict__ hist, const float* __restrict__ part,
                        float* __restrict__ dout) {
    __shared__ float red[256];
    int tid = threadIdx.x;
    float s = 0.f;
    for (int k = tid; k < KCODES; k += 256) {
        float pv = (float)hist[k] * (1.0f / 32768.0f);
        s += pv * logf(pv + 1e-10f);
    }
    red[tid] = s; __syncthreads();
    for (int st = 128; st > 0; st >>= 1) { if (tid < st) red[tid] += red[tid + st]; __syncthreads(); }
    if (tid == 0) {
        dout[PERP_OFF] = expf(-red[0]);
        float sse = 0.f;
        for (int i = 0; i < 512; i++) sse += part[i];
        dout[0] = 1.25f * sse / (float)((size_t)N_PTS * CDIM);
    }
}

extern "C" void kernel_launch(void* const* d_in, const int* in_sizes, int n_in,
                              void* d_out, int out_size, void* d_ws, size_t ws_size,
                              hipStream_t stream) {
    const float* x   = (const float*)d_in[0];
    const float* Wl  = (const float*)d_in[1];
    const float* bl  = (const float*)d_in[2];
    const float* emb = (const float*)d_in[3];
    float* out = (float*)d_out;

    char* ws = (char*)d_ws;
    float*  flat = (float*) (ws + 0);          // 16 MB
    __bf16* epk  = (__bf16*)(ws + 16777216);   // 4 MB, interleaved hi/lo B-fragments
    float*  hn   = (float*) (ws + 20971520);   // 32 KB
    int*    idx  = (int*)   (ws + 21004288);   // 128 KB
    int*    hist = (int*)   (ws + 21135360);   // 32 KB
    float*  part = (float*) (ws + 21168128);   // 2 KB

    hipLaunchKernelGGL(k_prep_emb, dim3(256), dim3(256), 0, stream, emb, epk);
    hipLaunchKernelGGL(k_norm,     dim3(32),  dim3(256), 0, stream, emb, hn);
    hipLaunchKernelGGL(k_zero,     dim3(32),  dim3(256), 0, stream, hist);
    hipLaunchKernelGGL(k_linear,   dim3(512), dim3(256), 0, stream, x, Wl, bl, flat);
    hipLaunchKernelGGL(k_argmin,   dim3(256), dim3(256), 0, stream, flat, epk, hn,
                       idx, out + IDX_OFF);
    hipLaunchKernelGGL(k_gather,   dim3(512), dim3(256), 0, stream, flat, emb, idx,
                       out + 1, hist, part);
    hipLaunchKernelGGL(k_final,    dim3(1),   dim3(256), 0, stream, hist, part, out);
}
